// SoftUnifAttention_37950331028198
// MI455X (gfx1250) — matrix operand order repaired and checked by reading, not compiled, not run on hardware
//
#include <hip/hip_runtime.h>
#include <hip/hip_bf16.h>
#include <math.h>
#include <stdint.h>

// Problem constants (match reference)
#define B_   2
#define S_   2048
#define HID_ 1024
#define NH_  16
#define HD_  64
#define VD_  32
#define TD_  32
#define EPS_ 1e-6f

typedef __attribute__((ext_vector_type(16))) _Float16 v16h;
typedef __attribute__((ext_vector_type(8)))  _Float16 h8;
typedef __attribute__((ext_vector_type(8)))  float    v8f;

// Assemble a 16-half WMMA operand fragment from two contiguous 8-half groups.
__device__ __forceinline__ v16h load_frag(const _Float16* p0, const _Float16* p1) {
    h8 g0 = *reinterpret_cast<const h8*>(p0);
    h8 g1 = *reinterpret_cast<const h8*>(p1);
    v16h a;
#pragma unroll
    for (int i = 0; i < 8; ++i) { a[i] = g0[i]; a[i + 8] = g1[i]; }
    return a;
}

__device__ __forceinline__ v8f wmma_f16(v16h a, v16h b, v8f c) {
    return __builtin_amdgcn_wmma_f32_16x16x32_f16(false, a, false, b, (short)0, c,
                                                  false, false);
}

// ---- CDNA5 async global->LDS copy (ASYNCcnt-tracked; CDNA5 ISA) ----
// 16B per lane: LDS byte address in a VGPR, 64-bit global address in a VGPR pair.
__device__ __forceinline__ void async_ld_b128(uint32_t lds_byte_off, const void* g) {
    asm volatile("global_load_async_to_lds_b128 %0, %1, off"
                 :: "v"(lds_byte_off), "v"((uint64_t)(uintptr_t)g)
                 : "memory");
}
__device__ __forceinline__ void wait_async0() {
    asm volatile("s_wait_asynccnt 0x0" ::: "memory");
}

// ---------------- conversion kernels ----------------

__global__ void cvt_f16_kernel(const float* __restrict__ in, _Float16* __restrict__ out, int n) {
    int i = blockIdx.x * blockDim.x + threadIdx.x;
    if (i < n) out[i] = (_Float16)in[i];
}

// f32 [K,N] -> f16 [N,K]   (transpose so B-fragment loads are contiguous)
__global__ void cvt_t_kernel(const float* __restrict__ in, _Float16* __restrict__ out,
                             int K, int N) {
    int i = blockIdx.x * blockDim.x + threadIdx.x;
    if (i >= K * N) return;
    int k = i / N, n = i % N;
    out[n * K + k] = (_Float16)in[i];
}

// bilinear (h,d,e) f32 -> bilT (h,e,d) f16
__global__ void cvt_bilT_kernel(const float* __restrict__ in, _Float16* __restrict__ out) {
    int i = blockIdx.x * blockDim.x + threadIdx.x;
    if (i >= NH_ * TD_ * TD_) return;
    int h = i / (TD_ * TD_);
    int rem = i % (TD_ * TD_);
    int d = rem / TD_, e = rem % TD_;
    out[(h * TD_ + e) * TD_ + d] = (_Float16)in[i];
}

// ---------------- generic WMMA GEMM: Y[M,N] = A[M,K] @ W[K,N] + bias ----------------
// A: f16 row-major [M,K]; WT: f16 transposed [N,K].
// Block = 8 waves sharing one 64-column W strip: the 64x32 W k-slice is staged in LDS
// via double-buffered async copies (one B128 chunk per thread per k-step) so the 8x
// redundant W traffic collapses to one DMA per block and overlaps the WMMAs.

template <int F32OUT>
__global__ void gemm_wmma_kernel(const _Float16* __restrict__ A,
                                 const _Float16* __restrict__ WT,
                                 const float* __restrict__ bias,
                                 _Float16* __restrict__ Yh,
                                 float* __restrict__ Yf,
                                 int M, int N, int K) {
    __shared__ _Float16 smem[8192];               // [0,4096): W double buffer; full 16KB: epilogue
    const int tid  = threadIdx.x;                 // 0..255
    const int lane = tid & 31;
    const int wid  = tid >> 5;
    const int m0 = (blockIdx.y * 8 + wid) * 16;
    const int n0 = blockIdx.x * 64;
    const int lrow = lane & 15;
    const int kb0  = (lane & 16) ? 16 : 0;        // A-fragment element K base
    const int kh   = (lane & 16) ? 16 : 0;        // B-fragment element K base
    const int roff = (lane & 16) ? 8  : 0;        // C/D row offset
    const uint32_t sbase = (uint32_t)(uintptr_t)smem;

    // one B128 chunk per thread fills one 64x32 f16 W tile (4KB)
    const int wn_local = tid >> 2;                // 0..63
    const int wc4      = tid & 3;                 // 0..3 (8-half chunk in a 32-half row)
    auto issue_w = [&](int k0, int buf) {
        async_ld_b128(sbase + (buf * 2048 + wn_local * 32 + wc4 * 8) * 2,
                      WT + (n0 + wn_local) * K + k0 + wc4 * 8);
    };

    v8f acc[4];
#pragma unroll
    for (int c = 0; c < 4; ++c)
#pragma unroll
        for (int r = 0; r < 8; ++r) acc[c][r] = 0.f;

    issue_w(0, 0);
    for (int k0 = 0; k0 < K; k0 += 32) {
        const int cur = (k0 >> 5) & 1;
        wait_async0();                            // own copies done
        __syncthreads();                          // everyone's copies done / prev reads done
        if (k0 + 32 < K) issue_w(k0 + 32, cur ^ 1);

        const _Float16* ap = A + (m0 + lrow) * K + k0 + kb0;
        v16h afrag = load_frag(ap, ap + 8);
        const _Float16* sw = smem + cur * 2048;
#pragma unroll
        for (int c = 0; c < 4; ++c) {
            const _Float16* bp = sw + (c * 16 + lrow) * 32 + kh;
            v16h bfrag = load_frag(bp, bp + 8);
            acc[c] = wmma_f16(afrag, bfrag, acc[c]);
        }
    }

    if constexpr (F32OUT) {
#pragma unroll
        for (int c = 0; c < 4; ++c) {
            const int n = n0 + c * 16 + lrow;
            const float bv = bias[n];
#pragma unroll
            for (int r = 0; r < 8; ++r)
                Yf[(m0 + roff + r) * N + n] = acc[c][r] + bv;
        }
    } else {
        // transpose the 16x64 tile through a per-wave LDS region -> coalesced b128 stores
        __syncthreads();                          // all waves done reading W buffers
        _Float16* ep = smem + wid * 1024;         // 16 rows x 64 cols f16 = 2KB
#pragma unroll
        for (int c = 0; c < 4; ++c) {
            const float bv = bias[n0 + c * 16 + lrow];
#pragma unroll
            for (int r = 0; r < 8; ++r)
                ep[(roff + r) * 64 + c * 16 + lrow] = (_Float16)(acc[c][r] + bv);
        }
        // wave-internal LDS ordering: stores precede loads in the in-order DS pipe
        const int row = lane >> 1;                // 0..15
        const int hsel = lane & 1;                // cols [0,32) / [32,64)
        const _Float16* rp = ep + row * 64 + hsel * 32;
        _Float16* yp = Yh + (m0 + row) * N + n0 + hsel * 32;
#pragma unroll
        for (int c4 = 0; c4 < 4; ++c4)
            *reinterpret_cast<h8*>(yp + c4 * 8) = *reinterpret_cast<const h8*>(rp + c4 * 8);
    }
}

// ---------------- split projections into per-head layouts ----------------

__global__ void split_kernel(const _Float16* __restrict__ qf,
                             const _Float16* __restrict__ kf,
                             const _Float16* __restrict__ vf,
                             _Float16* __restrict__ q_val, _Float16* __restrict__ q_type,
                             _Float16* __restrict__ k_val, _Float16* __restrict__ k_type,
                             _Float16* __restrict__ vT) {
    int i = blockIdx.x * blockDim.x + threadIdx.x;
    if (i >= B_ * S_ * HID_) return;
    int token = i / HID_, col = i % HID_;
    int b = token / S_, s = token % S_;
    int h = col / HD_, d = col % HD_;
    int bh = b * NH_ + h;
    _Float16 qe = qf[i], ke = kf[i], ve = vf[i];
    if (d < VD_) {
        q_val[(bh * S_ + s) * VD_ + d] = qe;
        k_val[(bh * S_ + s) * VD_ + d] = ke;
    } else {
        q_type[(bh * S_ + s) * TD_ + (d - VD_)] = qe;
        k_type[(bh * S_ + s) * TD_ + (d - VD_)] = ke;
    }
    vT[(bh * HD_ + d) * S_ + s] = ve;
}

// ---------------- qt = q_type @ bilinear (per head), WMMA 16x16x32 ----------------

__global__ void qt_gemm_kernel(const _Float16* __restrict__ qtype,
                               const _Float16* __restrict__ bilT,
                               _Float16* __restrict__ qt) {
    const int lane = threadIdx.x & 31;
    const int wid  = threadIdx.x >> 5;
    const int tile = blockIdx.x * 8 + wid;        // over B_*NH_*(S_/16)
    const int tiles_per_bh = S_ / 16;
    const int bh = tile / tiles_per_bh;
    const int m0 = (tile % tiles_per_bh) * 16;
    const int h = bh % NH_;
    const int lrow = lane & 15;
    const int kb0  = (lane & 16) ? 16 : 0;
    const int kh   = (lane & 16) ? 16 : 0;
    const int roff = (lane & 16) ? 8  : 0;

    const _Float16* ap = qtype + (bh * S_ + m0 + lrow) * TD_ + kb0;
    v16h afrag = load_frag(ap, ap + 8);
#pragma unroll
    for (int c = 0; c < 2; ++c) {
        const int e = c * 16 + lrow;
        const _Float16* bp = bilT + (h * TD_ + e) * TD_ + kh;
        v16h bfrag = load_frag(bp, bp + 8);
        v8f accv;
#pragma unroll
        for (int r = 0; r < 8; ++r) accv[r] = 0.f;
        accv = wmma_f16(afrag, bfrag, accv);
#pragma unroll
        for (int r = 0; r < 8; ++r)
            qt[(bh * S_ + m0 + roff + r) * TD_ + e] = (_Float16)accv[r];
    }
}

// ---------------- fused attention (online softmax, transposed-score trick) ----------------
// One wave owns a 16-query strip; the 4 waves of a block share (b,h) and consume the SAME
// K/V tiles, so each 32-key block of k_val/k_type/vT (8KB) is staged in LDS via
// double-buffered async copies (4 B128 chunks per thread), overlapping DMA with WMMAs.

__global__ void attn_kernel(const _Float16* __restrict__ qv,
                            const _Float16* __restrict__ qtv,
                            const _Float16* __restrict__ kv,
                            const _Float16* __restrict__ ktv,
                            const _Float16* __restrict__ vT,
                            const int* __restrict__ mask,
                            const float* __restrict__ type_bias,
                            _Float16* __restrict__ ao) {
    __shared__ _Float16 asmem[8192];              // 2 x (1024 kval + 1024 ktyp + 2048 vT) halfs
    const int tid  = threadIdx.x;                 // 0..127
    const int lane = tid & 31;
    const int wid  = tid >> 5;
    const int bh = blockIdx.y;
    const int b = bh / NH_;
    const int h = bh % NH_;
    const int q0 = (blockIdx.x * 4 + wid) * 16;
    const int lrow = lane & 15;
    const int kb0  = (lane & 16) ? 16 : 0;
    const int kh   = (lane & 16) ? 16 : 0;
    const int roff = (lane & 16) ? 8  : 0;
    const float tb = type_bias[h];
    const float inv_sqrt_vd = 0.17677669529663687f;   // 1/sqrt(32)
    const uint32_t sbase = (uint32_t)(uintptr_t)asmem;

    const _Float16* kvg = kv  + bh * S_ * VD_;
    const _Float16* ktg = ktv + bh * S_ * TD_;
    const _Float16* vtg = vT  + bh * HD_ * S_;

    // 512 B128 chunks per 32-key block: 128 k_val, 128 k_type, 256 vT rows
    auto issue_tiles = [&](int kb, int buf) {
        const uint32_t kvl = sbase + (buf * 4096) * 2;
        const uint32_t ktl = kvl + 1024 * 2;
        const uint32_t vl  = kvl + 2048 * 2;
#pragma unroll
        for (int j = 0; j < 4; ++j) {
            const int idx = tid + j * 128;
            if (idx < 128) {
                async_ld_b128(kvl + idx * 16, kvg + kb * VD_ + idx * 8);
            } else if (idx < 256) {
                const int i2 = idx - 128;
                async_ld_b128(ktl + i2 * 16, ktg + kb * TD_ + i2 * 8);
            } else {
                const int i2 = idx - 256;
                const int dim = i2 >> 2, c4 = i2 & 3;
                async_ld_b128(vl + (dim * 32 + c4 * 8) * 2, vtg + dim * S_ + kb + c4 * 8);
            }
        }
    };

    // Q fragments (B-layout): lane = query column, elements = feature dims.
    const _Float16* qp  = qv  + (bh * S_ + q0 + lrow) * VD_ + kh;
    const _Float16* qtp = qtv + (bh * S_ + q0 + lrow) * TD_ + kh;
    v16h bq  = load_frag(qp,  qp  + 8);
    v16h bqt = load_frag(qtp, qtp + 8);

    float m_run = -3.0e38f, l_run = 0.f;
    v8f acc[4];
#pragma unroll
    for (int c = 0; c < 4; ++c)
#pragma unroll
        for (int r = 0; r < 8; ++r) acc[c][r] = 0.f;

    const int* mrow = mask + b * S_;

    issue_tiles(0, 0);
    for (int kb = 0; kb < S_; kb += 32) {
        const int cur = (kb >> 5) & 1;
        wait_async0();
        __syncthreads();
        if (kb + 32 < S_) issue_tiles(kb + 32, cur ^ 1);

        const _Float16* skv = asmem + cur * 4096;
        const _Float16* skt = skv + 1024;
        const _Float16* svv = skv + 2048;

        // K fragments (A-layout) from LDS: lane = key row, elements = feature dims.
        const _Float16* kp0 = skv + lrow * VD_ + kb0;
        const _Float16* kp1 = skv + (16 + lrow) * VD_ + kb0;
        const _Float16* tp0 = skt + lrow * TD_ + kb0;
        const _Float16* tp1 = skt + (16 + lrow) * TD_ + kb0;
        v16h ak0 = load_frag(kp0, kp0 + 8);
        v16h ak1 = load_frag(kp1, kp1 + 8);
        v16h at0 = load_frag(tp0, tp0 + 8);
        v16h at1 = load_frag(tp1, tp1 + 8);

        v8f z;
#pragma unroll
        for (int r = 0; r < 8; ++r) z[r] = 0.f;
        v8f D0 = wmma_f16(ak0, bq,  z);   // val scores^T, keys kb..kb+15
        v8f D1 = wmma_f16(ak1, bq,  z);   // keys kb+16..kb+31
        v8f E0 = wmma_f16(at0, bqt, z);   // type scores^T
        v8f E1 = wmma_f16(at1, bqt, z);

        float p0[8], p1[8];
        float mloc = -3.0e38f;
#pragma unroll
        for (int r = 0; r < 8; ++r) {
            int key0 = kb + roff + r;
            int key1 = key0 + 16;
            float z0 = E0[r] + tb;
            float s0 = D0[r] * inv_sqrt_vd + logf(1.f / (1.f + expf(-z0)) + EPS_);
            if (mrow[key0] == 0) s0 = -1.0e30f;
            float z1 = E1[r] + tb;
            float s1 = D1[r] * inv_sqrt_vd + logf(1.f / (1.f + expf(-z1)) + EPS_);
            if (mrow[key1] == 0) s1 = -1.0e30f;
            p0[r] = s0; p1[r] = s1;
            mloc = fmaxf(mloc, fmaxf(s0, s1));
        }
        // per-query max across the (lane, lane^16) pair
        float m_blk = fmaxf(mloc, __shfl_xor(mloc, 16, 32));
        float m_new = fmaxf(m_run, m_blk);
        float scl = expf(m_run - m_new);
#pragma unroll
        for (int c = 0; c < 4; ++c)
#pragma unroll
            for (int r = 0; r < 8; ++r) acc[c][r] *= scl;

        // exponentiate + pair-exchange into B-fragment layout (32 keys x 16 queries)
        float lsum = 0.f;
        v16h bp;
#pragma unroll
        for (int r = 0; r < 8; ++r) {
            p0[r] = expf(p0[r] - m_new);
            p1[r] = expf(p1[r] - m_new);
            float e0 = __shfl_xor(p0[r], 16, 32);
            float e1 = __shfl_xor(p1[r], 16, 32);
            float lo = (lane < 16) ? p0[r] : e1;
            float hi = (lane < 16) ? e0 : p1[r];
            bp[r]     = (_Float16)lo;
            bp[r + 8] = (_Float16)hi;
            lsum += lo + hi;
        }
        l_run = l_run * scl + lsum + __shfl_xor(lsum, 16, 32);
        m_run = m_new;

        // out^T += V^T(16 dims x 32 keys) @ P^T(32 keys x 16 queries), V^T from LDS
#pragma unroll
        for (int c = 0; c < 4; ++c) {
            const _Float16* vp = svv + (c * 16 + lrow) * 32 + kb0;
            v16h av = load_frag(vp, vp + 8);
            acc[c] = wmma_f16(av, bp, acc[c]);
        }
    }

    float rinv = (l_run > 0.f) ? 1.f / l_run : 0.f;
    const int token = b * S_ + q0 + lrow;   // lane = query
#pragma unroll
    for (int c = 0; c < 4; ++c) {
        h8 st;
#pragma unroll
        for (int r = 0; r < 8; ++r) st[r] = (_Float16)(acc[c][r] * rinv);
        *reinterpret_cast<h8*>(ao + token * HID_ + h * HD_ + c * 16 + roff) = st;
    }
}

// ---------------- host launcher ----------------

extern "C" void kernel_launch(void* const* d_in, const int* in_sizes, int n_in,
                              void* d_out, int out_size, void* d_ws, size_t ws_size,
                              hipStream_t stream) {
    const float* x     = (const float*)d_in[0];
    const float* q_w   = (const float*)d_in[1];
    const float* q_b   = (const float*)d_in[2];
    const float* k_w   = (const float*)d_in[3];
    const float* k_b   = (const float*)d_in[4];
    const float* v_w   = (const float*)d_in[5];
    const float* v_b   = (const float*)d_in[6];
    const float* out_w = (const float*)d_in[7];
    const float* out_b = (const float*)d_in[8];
    const float* bil   = (const float*)d_in[9];
    const float* tbias = (const float*)d_in[10];
    const int*   mask  = (const int*)d_in[11];
    float* outp = (float*)d_out;

    const int M = B_ * S_;                         // 4096 tokens
    char* ws = (char*)d_ws;
    size_t off = 0;
    auto take = [&](size_t bytes) -> void* {
        void* p = ws + off;
        off += (bytes + 255) & ~(size_t)255;
        return p;
    };
    _Float16* xh    = (_Float16*)take((size_t)M * HID_ * 2);
    _Float16* wqT   = (_Float16*)take((size_t)HID_ * HID_ * 2);
    _Float16* wkT   = (_Float16*)take((size_t)HID_ * HID_ * 2);
    _Float16* wvT   = (_Float16*)take((size_t)HID_ * HID_ * 2);
    _Float16* woT   = (_Float16*)take((size_t)HID_ * HID_ * 2);
    _Float16* bilT  = (_Float16*)take((size_t)NH_ * TD_ * TD_ * 2);
    _Float16* qfull = (_Float16*)take((size_t)M * HID_ * 2);
    _Float16* kfull = (_Float16*)take((size_t)M * HID_ * 2);
    _Float16* vfull = (_Float16*)take((size_t)M * HID_ * 2);
    _Float16* q_val = (_Float16*)take((size_t)B_ * NH_ * S_ * VD_ * 2);
    _Float16* q_typ = (_Float16*)take((size_t)B_ * NH_ * S_ * TD_ * 2);
    _Float16* k_val = (_Float16*)take((size_t)B_ * NH_ * S_ * VD_ * 2);
    _Float16* k_typ = (_Float16*)take((size_t)B_ * NH_ * S_ * TD_ * 2);
    _Float16* vT    = (_Float16*)take((size_t)B_ * NH_ * HD_ * S_ * 2);
    _Float16* qt    = (_Float16*)take((size_t)B_ * NH_ * S_ * TD_ * 2);
    _Float16* ao    = xh;   // x is dead after the Q/K/V projections; reuse its buffer

    const int nElemX = M * HID_;
    const int nElemW = HID_ * HID_;

    cvt_f16_kernel<<<(nElemX + 255) / 256, 256, 0, stream>>>(x, xh, nElemX);
    cvt_t_kernel<<<(nElemW + 255) / 256, 256, 0, stream>>>(q_w,   wqT, HID_, HID_);
    cvt_t_kernel<<<(nElemW + 255) / 256, 256, 0, stream>>>(k_w,   wkT, HID_, HID_);
    cvt_t_kernel<<<(nElemW + 255) / 256, 256, 0, stream>>>(v_w,   wvT, HID_, HID_);
    cvt_t_kernel<<<(nElemW + 255) / 256, 256, 0, stream>>>(out_w, woT, HID_, HID_);
    cvt_bilT_kernel<<<(NH_ * TD_ * TD_ + 255) / 256, 256, 0, stream>>>(bil, bilT);

    dim3 ggrid(HID_ / 64, M / (16 * 8));           // (16, 32), 256 threads = 8 waves
    gemm_wmma_kernel<0><<<ggrid, 256, 0, stream>>>(xh, wqT, q_b, qfull, nullptr, M, HID_, HID_);
    gemm_wmma_kernel<0><<<ggrid, 256, 0, stream>>>(xh, wkT, k_b, kfull, nullptr, M, HID_, HID_);
    gemm_wmma_kernel<0><<<ggrid, 256, 0, stream>>>(xh, wvT, v_b, vfull, nullptr, M, HID_, HID_);

    split_kernel<<<(nElemX + 255) / 256, 256, 0, stream>>>(qfull, kfull, vfull,
                                                           q_val, q_typ, k_val, k_typ, vT);

    qt_gemm_kernel<<<B_ * NH_ * (S_ / 16) / 8, 256, 0, stream>>>(q_typ, bilT, qt);

    dim3 agrid(S_ / 16 / 4, B_ * NH_);             // (32, 32), 128 threads = 4 waves
    attn_kernel<<<agrid, 128, 0, stream>>>(q_val, qt, k_val, k_typ, vT, mask, tbias, ao);

    gemm_wmma_kernel<1><<<ggrid, 256, 0, stream>>>(ao, woT, out_b, nullptr, outp, M, HID_, HID_);
}
